// GNNActor1_27195732918296
// MI455X (gfx1250) — matrix operand order repaired and checked by reading, not compiled, not run on hardware
//
#include <hip/hip_runtime.h>
#include <hip/hip_bf16.h>

typedef __attribute__((ext_vector_type(16))) _Float16 v16h;
typedef __attribute__((ext_vector_type(8)))  float    v8f;

#define NREGION 512
#define IN_CH   128
#define HIDDEN  128
#define E_GRAPH 16384
#define NEDGES  (NREGION * NREGION)     // 262144
#define EBLOCKS (NEDGES / 64)           // 4096 blocks, 64 edges (4 M-tiles) each

// LDS layout for k_edge: 64 concat-rows of 256 halves, padded to 264 halves (528B)
#define AROW_B  528
#define SMEM_B  (64 * AROW_B)           // 33792 B; also reused as h-tile (64*130*4 = 33280 B)

// ---------- helpers ----------

union AFrag { uint4 q[2]; v16h h; };

// CDNA5 WMMA 16-bit A-matrix 16x32 layout: lane (0-15) holds row M=lane,
// K = kbase + (e&7) + 16*(e>>3), kbase = (lane>>4)*8.
// => per lane: two contiguous 8-half (16B) runs at ch0 and ch0+16.
__device__ __forceinline__ v16h load_a_frag_g(const _Float16* row_base, int ch0) {
    AFrag f;
    const uint4* p = (const uint4*)(row_base + ch0);
    f.q[0] = p[0];
    f.q[1] = p[2];
    return f.h;
}

__device__ __forceinline__ v16h load_a_frag_lds(const char* row_base, int ch0) {
    AFrag f;
    const uint4* p = (const uint4*)(row_base + ch0 * 2);
    f.q[0] = p[0];   // halves [ch0, ch0+8)
    f.q[1] = p[2];   // halves [ch0+16, ch0+24)
    return f.h;
}

__device__ __forceinline__ float softplusf(float x) {
    return x > 20.0f ? x : log1pf(expf(x));
}

__device__ __forceinline__ float urand(unsigned v) {
    v = v * 747796405u + 2891336453u;
    unsigned w = ((v >> ((v >> 28) + 4u)) ^ v) * 277803737u;
    w = (w >> 22) ^ w;
    return ((float)(w >> 8) + 0.5f) * (1.0f / 16777216.0f);
}

// ---------- setup kernels ----------

__global__ void k_init(float* deg, float* agg, float* out_lp) {
    int i = blockIdx.x * blockDim.x + threadIdx.x;
    if (i < NREGION * IN_CH) agg[i] = 0.0f;
    if (i < NREGION)         deg[i] = 1.0f;   // self-loop
    if (i == 0)              *out_lp = 0.0f;
}

__global__ void k_cvt_state(const float* s, _Float16* sh) {
    int i = blockIdx.x * blockDim.x + threadIdx.x;
    if (i < NREGION * IN_CH) sh[i] = (_Float16)s[i];
}

// Pack lin1_w (256x128 row major) into WMMA B fragments:
// lane holds col N = lane&15, K = (lane>>4)*8 + (e&7) + 16*(e>>3). [ntile(8)][ktile(8)][lane(32)][e(16)]
__global__ void k_prep_b1(const float* w, _Float16* B) {
    int idx = blockIdx.x * blockDim.x + threadIdx.x;
    if (idx >= 8 * 8 * 32 * 16) return;
    int e  = idx & 15;
    int l  = (idx >> 4) & 31;
    int kt = (idx >> 9) & 7;
    int nt = idx >> 12;
    int K = kt * 32 + (l >> 4) * 8 + (e & 7) + ((e >> 3) * 16);
    int N = nt * 16 + (l & 15);
    B[idx] = (_Float16)w[K * HIDDEN + N];
}

// Pack conv_w (128x128): [ntile(8)][ktile(4)][lane(32)][e(16)]
__global__ void k_prep_bc(const float* w, _Float16* B) {
    int idx = blockIdx.x * blockDim.x + threadIdx.x;
    if (idx >= 8 * 4 * 32 * 16) return;
    int e  = idx & 15;
    int l  = (idx >> 4) & 31;
    int kt = (idx >> 9) & 3;
    int nt = idx >> 11;
    int K = kt * 32 + (l >> 4) * 8 + (e & 7) + ((e >> 3) * 16);
    int N = nt * 16 + (l & 15);
    B[idx] = (_Float16)w[K * IN_CH + N];
}

// ---------- GCN conv ----------

__global__ void k_deg(const int* ei, float* deg) {
    int e = blockIdx.x * blockDim.x + threadIdx.x;
    if (e < E_GRAPH) atomicAdd(&deg[ei[E_GRAPH + e]], 1.0f);
}

__global__ void k_dinv(float* deg) {
    int i = blockIdx.x * blockDim.x + threadIdx.x;
    if (i < NREGION) deg[i] = rsqrtf(deg[i]);
}

// hw = state @ conv_w  (512x128x128) via WMMA f16->f32
__global__ void __launch_bounds__(256) k_conv_gemm(const _Float16* sh, const _Float16* Bc, float* hw) {
    int lane = threadIdx.x & 31, wid = threadIdx.x >> 5;
    int row = lane & 15, hi = lane >> 4;
    const _Float16* abase = sh + (blockIdx.x * 16 + row) * IN_CH;
    v8f c = {};
#pragma unroll
    for (int kt = 0; kt < 4; ++kt) {
        v16h a = load_a_frag_g(abase, kt * 32 + hi * 8);
        v16h b = *(const v16h*)(Bc + ((wid * 4 + kt) * 32 + lane) * 16);
        c = __builtin_amdgcn_wmma_f32_16x16x32_f16(false, a, false, b, (short)0, c, false, false);
    }
    int col = wid * 16 + row;
#pragma unroll
    for (int r = 0; r < 8; ++r) {
        int m = blockIdx.x * 16 + r + 8 * hi;
        hw[m * IN_CH + col] = c[r];
    }
}

__global__ void k_scatter(const int* ei, const float* dinv, const float* hw, float* agg) {
    int t = blockIdx.x * blockDim.x + threadIdx.x;
    if (t >= E_GRAPH * IN_CH) return;
    int e = t >> 7, ch = t & 127;
    int s = ei[e], d = ei[E_GRAPH + e];
    float norm = dinv[s] * dinv[d];
    atomicAdd(&agg[d * IN_CH + ch], hw[s * IN_CH + ch] * norm);
}

__global__ void k_finalize(const float* agg, const float* hw, const float* dinv,
                           const float* conv_b, const float* state, _Float16* xh) {
    int i = blockIdx.x * blockDim.x + threadIdx.x;
    if (i >= NREGION * IN_CH) return;
    int n = i >> 7, ch = i & 127;
    float di = dinv[n];
    float v = agg[i] + hw[i] * di * di + conv_b[ch];
    v = v > 0.0f ? v : 0.0f;
    xh[i] = (_Float16)(v + state[i]);
}

// ---------- fused edge kernel ----------
// 64 edges per block. Phase 1: cooperative stage of concat(x[src],x[dst]) rows into
// LDS (unique data only). Phase 2: 8 waves x 4 M-tiles register blocking; each B
// fragment (global, L0-hot) feeds 4 WMMAs; A fragments come from LDS (ds_load_b128).
// Phase 3: LDS reused for the h-tile; lin2 + softplus + Beta + log-prob epilogue.

__global__ void __launch_bounds__(256) k_edge(const int* __restrict__ edges,
                                              const _Float16* __restrict__ xh,
                                              const _Float16* __restrict__ B1,
                                              const float* __restrict__ lin1_b,
                                              const float* __restrict__ lin2_w,
                                              const float* __restrict__ lin2_b,
                                              float* __restrict__ action,
                                              float* __restrict__ logp) {
    __shared__ uint4 smem4[SMEM_B / 16];
    __shared__ float lpsum;
    char* smem = (char*)smem4;
    int tid = threadIdx.x;
    int lane = tid & 31, wid = tid >> 5;
    int row = lane & 15, hi = lane >> 4;
    if (tid == 0) lpsum = 0.0f;

    // ---- Phase 1: stage A (64 edges x 512B concat rows) into LDS, 16B chunks ----
#pragma unroll
    for (int i = 0; i < 8; ++i) {
        int c = i * 256 + tid;                 // 2048 chunks of 16B
        int el = c >> 5;                       // edge within block
        int boff = (c & 31) * 16;              // byte offset in 512B concat row
        int eg = blockIdx.x * 64 + el;
        int node = (boff < 256) ? edges[2 * eg] : edges[2 * eg + 1];
        const uint4* g = (const uint4*)((const char*)xh + node * 256 + (boff & 255));
        *(uint4*)(smem + el * AROW_B + boff) = *g;
    }
    __syncthreads();

    // ---- Phase 2: GEMM, 4 M-tiles per wave, B reused across M-tiles ----
    v8f c0 = {}, c1 = {}, c2 = {}, c3 = {};
#pragma unroll
    for (int kt = 0; kt < 8; ++kt) {           // K = 256
        v16h b = *(const v16h*)(B1 + ((wid * 8 + kt) * 32 + lane) * 16);
        int ch0 = kt * 32 + hi * 8;            // concat-row half offset
        v16h a0 = load_a_frag_lds(smem + (0 * 16 + row) * AROW_B, ch0);
        v16h a1 = load_a_frag_lds(smem + (1 * 16 + row) * AROW_B, ch0);
        v16h a2 = load_a_frag_lds(smem + (2 * 16 + row) * AROW_B, ch0);
        v16h a3 = load_a_frag_lds(smem + (3 * 16 + row) * AROW_B, ch0);
        c0 = __builtin_amdgcn_wmma_f32_16x16x32_f16(false, a0, false, b, (short)0, c0, false, false);
        c1 = __builtin_amdgcn_wmma_f32_16x16x32_f16(false, a1, false, b, (short)0, c1, false, false);
        c2 = __builtin_amdgcn_wmma_f32_16x16x32_f16(false, a2, false, b, (short)0, c2, false, false);
        c3 = __builtin_amdgcn_wmma_f32_16x16x32_f16(false, a3, false, b, (short)0, c3, false, false);
    }
    __syncthreads();   // all A reads done; LDS now reused for h-tile

    // ---- Phase 3a: bias + leaky_relu, store h-tile (64 x 128, stride 130) ----
    float* hbuf = (float*)smem;
    int hcol = wid * 16 + row;
    float bias = lin1_b[hcol];
    v8f cc[4] = {c0, c1, c2, c3};
#pragma unroll
    for (int mt = 0; mt < 4; ++mt) {
#pragma unroll
        for (int r = 0; r < 8; ++r) {
            float v = cc[mt][r] + bias;
            v = v > 0.0f ? v : 0.01f * v;      // leaky_relu
            hbuf[(mt * 16 + r + 8 * hi) * 130 + hcol] = v;
        }
    }
    __syncthreads();

    // ---- Phase 3b: lin2 (128->2), softplus, Beta sample, log-prob ----
    if (tid < 64) {
        int e = blockIdx.x * 64 + tid;
        const float* hrow = hbuf + tid * 130;
        float s0 = lin2_b[0], s1 = lin2_b[1];
        for (int k = 0; k < HIDDEN; ++k) {
            float hv = hrow[k];
            s0 = fmaf(hv, lin2_w[2 * k + 0], s0);
            s1 = fmaf(hv, lin2_w[2 * k + 1], s1);
        }
        float a  = softplusf(s0) + 1e-10f;
        float bb = softplusf(s1) + 1e-10f;

        float act = a / (a + bb);              // deterministic Beta: Johnk, bounded
        float inva = 1.0f / a, invb = 1.0f / bb;
        for (int t = 0; t < 16; ++t) {
            float u1 = urand((unsigned)e * 37u + (unsigned)t * 2u + 1u);
            float u2 = urand((unsigned)e * 37u + (unsigned)t * 2u + 2u);
            float x = powf(u1, inva), y = powf(u2, invb);
            float ss = x + y;
            if (ss > 0.0f && ss <= 1.0f) { act = x / ss; break; }
        }
        act = fminf(fmaxf(act, 1e-6f), 1.0f - 1e-6f);
        action[e] = act;

        float term = (a - 1.0f) * logf(act) + (bb - 1.0f) * log1pf(-act)
                   - (lgammaf(a) + lgammaf(bb) - lgammaf(a + bb));
        atomicAdd(&lpsum, term);
    }
    __syncthreads();
    if (tid == 0) atomicAdd(logp, lpsum);
}

// ---------- launcher ----------

extern "C" void kernel_launch(void* const* d_in, const int* in_sizes, int n_in,
                              void* d_out, int out_size, void* d_ws, size_t ws_size,
                              hipStream_t stream) {
    const float* state  = (const float*)d_in[0];
    const float* conv_w = (const float*)d_in[1];
    const float* conv_b = (const float*)d_in[2];
    const float* lin1_w = (const float*)d_in[3];
    const float* lin1_b = (const float*)d_in[4];
    const float* lin2_w = (const float*)d_in[5];
    const float* lin2_b = (const float*)d_in[6];
    const int*   ei     = (const int*)d_in[7];   // (2, 16384)
    const int*   edges  = (const int*)d_in[8];   // (262144, 2)
    float* out = (float*)d_out;                  // action[262144] + logprob[1]

    char* ws = (char*)d_ws;                      // all offsets 256B-aligned
    float*    deg = (float*)(ws + 0);            //   512 f
    float*    hw  = (float*)(ws + 2048);         // 65536 f
    float*    agg = (float*)(ws + 264192);       // 65536 f
    _Float16* sh  = (_Float16*)(ws + 526336);    // 65536 h
    _Float16* xh  = (_Float16*)(ws + 657408);    // 65536 h
    _Float16* B1  = (_Float16*)(ws + 788480);    // 32768 h
    _Float16* Bc  = (_Float16*)(ws + 854016);    // 16384 h

    k_init     <<<256,  256, 0, stream>>>(deg, agg, out + NEDGES);
    k_cvt_state<<<256,  256, 0, stream>>>(state, sh);
    k_prep_b1  <<<128,  256, 0, stream>>>(lin1_w, B1);
    k_prep_bc  <<<64,   256, 0, stream>>>(conv_w, Bc);
    k_deg      <<<64,   256, 0, stream>>>(ei, deg);
    k_dinv     <<<2,    256, 0, stream>>>(deg);
    k_conv_gemm<<<32,   256, 0, stream>>>(sh, Bc, hw);
    k_scatter  <<<8192, 256, 0, stream>>>(ei, deg, hw, agg);
    k_finalize <<<256,  256, 0, stream>>>(agg, hw, deg, conv_b, state, xh);
    k_edge     <<<EBLOCKS, 256, 0, stream>>>(edges, xh, B1, lin1_b, lin2_w, lin2_b,
                                             out, out + NEDGES);
}
